// PathInstanceNetwork_60206851555986
// MI455X (gfx1250) — compile-verified, matrix operand order checked by reading, not run on hardware
//
#include <hip/hip_runtime.h>
#include <hip/hip_bf16.h>

// Problem constants (reference):
// B=16, P=16, I=16, L=64, D=128, F=128, C=1024
#define BB 16
#define PP 16
#define II 16
#define LL 64
#define DD 128
#define FF 128
#define CC 1024
#define NEGV (-1000000000.0f)

typedef __attribute__((ext_vector_type(16))) __bf16 v16bf;
typedef __attribute__((ext_vector_type(8)))  __bf16 v8bf;
typedef __attribute__((ext_vector_type(8)))  float  v8f;

static __device__ __forceinline__ __bf16 f2bf(float f) {
    unsigned u = __builtin_bit_cast(unsigned, f);
    unsigned r = u + 0x7FFFu + ((u >> 16) & 1u);   // round-to-nearest-even
    unsigned short h = (unsigned short)(r >> 16);
    return __builtin_bit_cast(__bf16, h);
}

static __device__ __forceinline__ __bf16 bf_zero() {
    return __builtin_bit_cast(__bf16, (unsigned short)0);
}

// Load one 16x32 WMMA operand tile from row-major bf16 data.
// Per ISA 7.12.2 the per-lane K set is {0..7, 16..23} + 8*half (+kbase):
// exactly two contiguous 16-byte chunks -> two b128 loads, zero VALU.
static __device__ __forceinline__ v16bf load_row(const __bf16* rowp, int kbase) {
    int half = (threadIdx.x >> 4) & 1;
    const __bf16* p = rowp + kbase + 8 * half;
    v8bf lo = *(const v8bf*)(p);
    v8bf hi = *(const v8bf*)(p + 16);
    return __builtin_shufflevector(lo, hi, 0, 1, 2, 3, 4, 5, 6, 7,
                                   8, 9, 10, 11, 12, 13, 14, 15);
}

static __device__ __forceinline__ v8f wmma_bf16(v16bf a, v16bf b, v8f c) {
    return __builtin_amdgcn_wmma_f32_16x16x32_bf16(false, a, false, b,
                                                   (short)0, c, false, false);
}

// ---------------------------------------------------------------------------
// K0: one-shot f32 -> bf16 weight conversion into workspace
// ---------------------------------------------------------------------------
__global__ __launch_bounds__(256) void k_cvt(const float* __restrict__ src,
                                             __bf16* __restrict__ dst, int n) {
    int i = blockIdx.x * 256 + threadIdx.x;
    if (i < n) dst[i] = f2bf(src[i]);
}

// ---------------------------------------------------------------------------
// K1: masked mean over L.  mean_x[b,p,i,d] = sum_{l<len} path_emb / len
// One block per (b,p,i); 256 threads = 2 rows in flight, coalesced 512B rows.
// ---------------------------------------------------------------------------
__global__ __launch_bounds__(256) void k_mean(const float* __restrict__ pe,
                                              const int* __restrict__ plen,
                                              float* __restrict__ mean_x) {
    int idx = blockIdx.x;              // 0..4095
    int t = threadIdx.x;
    int d = t & 127;
    int half = t >> 7;                 // 0 or 1
    int len = plen[idx];
    const float* src = pe + (size_t)idx * LL * DD;
    float acc = 0.0f;
    for (int l = half; l < len; l += 2) acc += src[l * DD + d];
    __shared__ float s[256];
    s[t] = acc;
    __syncthreads();
    if (t < 128)
        mean_x[(size_t)idx * DD + t] = (s[t] + s[t + 128]) / (float)len;
}

// ---------------------------------------------------------------------------
// K2: inst_bf = bf16( mean_x(4096x128) @ conv_w^T + conv_b )  (WMMA bf16)
// ---------------------------------------------------------------------------
__global__ __launch_bounds__(256) void k_inst(const float* __restrict__ mean_x,
                                              const __bf16* __restrict__ conv_w_bf,
                                              const float* __restrict__ conv_b,
                                              __bf16* __restrict__ inst_bf) {
    __shared__ __attribute__((aligned(16))) __bf16 sx[16 * DD];
    int t = threadIdx.x;
    int rt = blockIdx.x;               // 0..255
    const float* src = mean_x + (size_t)rt * 16 * DD;
    for (int i = t; i < 16 * DD; i += 256) sx[i] = f2bf(src[i]);
    __syncthreads();

    int w = t >> 5, lane = t & 31, lr = lane & 15, half = lane >> 4;
    const __bf16* arow = sx + lr * DD;
    const __bf16* brow = conv_w_bf + (w * 16 + lr) * DD;
    v8f acc = {};
#pragma unroll
    for (int kc = 0; kc < 4; ++kc)
        acc = wmma_bf16(load_row(arow, kc * 32), load_row(brow, kc * 32), acc);
    int col = w * 16 + lr;
    float bias = conv_b[col];
    __bf16* drow = inst_bf + (size_t)rt * 16 * FF;
#pragma unroll
    for (int j = 0; j < 8; ++j) {
        int m = j + 8 * half;
        drow[m * FF + col] = f2bf(acc[j] + bias);
    }
}

// ---------------------------------------------------------------------------
// K3: per-(b,p) single-head attention over I=16, mean over I -> meta_bf.
// One workgroup (8 waves) per (b,p). All operands staged as bf16 in LDS.
// ---------------------------------------------------------------------------
__global__ __launch_bounds__(256) void k_pia(const __bf16* __restrict__ inst_bf,
                                             const __bf16* __restrict__ in_w_bf,
                                             const float* __restrict__ in_b,
                                             const __bf16* __restrict__ out_w_bf,
                                             const float* __restrict__ out_b,
                                             __bf16* __restrict__ meta_bf) {
    __shared__ __attribute__((aligned(16))) __bf16 sx[16 * FF];
    __shared__ __attribute__((aligned(16))) __bf16 qb[16 * FF];
    __shared__ __attribute__((aligned(16))) __bf16 kb[16 * FF];
    __shared__ __attribute__((aligned(16))) __bf16 vT[FF * 32];   // [f][j], j>=16 zero
    __shared__ __attribute__((aligned(16))) __bf16 attn_bf[16 * 32];
    __shared__ __attribute__((aligned(16))) float  sattn[16 * 16];
    __shared__ __attribute__((aligned(16))) __bf16 sco[16 * FF];
    int t = threadIdx.x, w = t >> 5, lane = t & 31, lr = lane & 15, half = lane >> 4;
    size_t bp = blockIdx.x;

    // stage inst tile (bf16, 4KB) as uint4 copies; zero vT's j-pad
    ((uint4*)sx)[t] = ((const uint4*)(inst_bf + bp * 16 * FF))[t];
    for (int i = t; i < FF * 16; i += 256)
        vT[(i >> 4) * 32 + 16 + (i & 15)] = bf_zero();
    __syncthreads();

    // qkv: 24 output tiles (16x384), K=128
    for (int ct = w; ct < 24; ct += 8) {
        v8f acc = {};
        const __bf16* ar = sx + lr * FF;
        const __bf16* br = in_w_bf + (ct * 16 + lr) * FF;
#pragma unroll
        for (int kc = 0; kc < 4; ++kc)
            acc = wmma_bf16(load_row(ar, kc * 32), load_row(br, kc * 32), acc);
        int col16 = (ct & 7) * 16 + lr;
        float bias = in_b[ct * 16 + lr];
        if (ct < 8) {          // q rows
#pragma unroll
            for (int j = 0; j < 8; ++j) {
                int m = j + 8 * half;
                qb[m * FF + col16] = f2bf(acc[j] + bias);
            }
        } else if (ct < 16) {  // k rows
#pragma unroll
            for (int j = 0; j < 8; ++j) {
                int m = j + 8 * half;
                kb[m * FF + col16] = f2bf(acc[j] + bias);
            }
        } else {               // v stored transposed: vT[f][j]
#pragma unroll
            for (int j = 0; j < 8; ++j) {
                int m = j + 8 * half;
                vT[col16 * 32 + m] = f2bf(acc[j] + bias);
            }
        }
    }
    __syncthreads();

    // S = q k^T / sqrt(F): single 16x16 tile, wave 0
    if (w == 0) {
        v8f acc = {};
        const __bf16* ar = qb + lr * FF;
        const __bf16* br = kb + lr * FF;   // B column n = k row n
#pragma unroll
        for (int kc = 0; kc < 4; ++kc)
            acc = wmma_bf16(load_row(ar, kc * 32), load_row(br, kc * 32), acc);
        const float scale = 0.08838834764831845f; // 1/sqrt(128)
#pragma unroll
        for (int j = 0; j < 8; ++j) {
            int m = j + 8 * half;
            sattn[m * 16 + lr] = acc[j] * scale;
        }
    }
    __syncthreads();

    // row softmax over j; write zero-padded bf16 probabilities
    if (t < 16) {
        float mx = -1e30f;
        for (int j = 0; j < 16; ++j) mx = fmaxf(mx, sattn[t * 16 + j]);
        float e[16], s = 0.0f;
        for (int j = 0; j < 16; ++j) { e[j] = __expf(sattn[t * 16 + j] - mx); s += e[j]; }
        float inv = 1.0f / s;
        for (int j = 0; j < 16; ++j) {
            attn_bf[t * 32 + j] = f2bf(e[j] * inv);
            attn_bf[t * 32 + 16 + j] = bf_zero();
        }
    }
    __syncthreads();

    // core = attn(16x16,padded K=32) @ v: wave w -> col tile w; B from vT rows
    {
        int ct = w;
        v8f acc = {};
        acc = wmma_bf16(load_row(attn_bf + lr * 32, 0),
                        load_row(vT + (ct * 16 + lr) * 32, 0), acc);
#pragma unroll
        for (int j = 0; j < 8; ++j) {
            int m = j + 8 * half;
            sco[m * FF + ct * 16 + lr] = f2bf(acc[j]);
        }
    }
    __syncthreads();

    // out = core @ out_w^T + out_b; mean over 16 rows -> meta_bf[b,p,:]
    {
        int ct = w;
        v8f acc = {};
        const __bf16* ar = sco + lr * FF;
        const __bf16* br = out_w_bf + (ct * 16 + lr) * FF;
#pragma unroll
        for (int kc = 0; kc < 4; ++kc)
            acc = wmma_bf16(load_row(ar, kc * 32), load_row(br, kc * 32), acc);
        float ps = 0.0f;
#pragma unroll
        for (int j = 0; j < 8; ++j) ps += acc[j];   // rows 8*half..8*half+7
        ps += __shfl_xor(ps, 16, 32);               // combine halves
        if (half == 0) {
            int col = ct * 16 + lr;
            meta_bf[bp * FF + col] = f2bf(ps * (1.0f / 16.0f) + out_b[col]);
        }
    }
}

// ---------------------------------------------------------------------------
// K4: per-b attention over P=16 (mpa) + qia value-path; folds query/v1/v2
// against score_w into the per-b scalar base[b].
// ---------------------------------------------------------------------------
__global__ __launch_bounds__(256) void k_mpa(const __bf16* __restrict__ meta_bf,
                                             const float* __restrict__ query,
                                             const __bf16* __restrict__ m_in_w_bf,
                                             const float* __restrict__ m_in_b,
                                             const __bf16* __restrict__ m_out_w_bf,
                                             const float* __restrict__ m_out_b,
                                             const __bf16* __restrict__ q_in_w_bf,
                                             const float* __restrict__ q_in_b,
                                             const __bf16* __restrict__ q_out_w_bf,
                                             const float* __restrict__ q_out_b,
                                             const float* __restrict__ score_w,
                                             const float* __restrict__ score_b,
                                             float* __restrict__ base_out) {
    __shared__ __attribute__((aligned(16))) __bf16 sx[16 * FF];
    __shared__ __attribute__((aligned(16))) __bf16 qb[16 * FF];
    __shared__ __attribute__((aligned(16))) __bf16 kb[16 * FF];
    __shared__ __attribute__((aligned(16))) __bf16 vT[FF * 32];
    __shared__ __attribute__((aligned(16))) __bf16 st[16 * FF];   // qia value path
    __shared__ __attribute__((aligned(16))) __bf16 attn_bf[16 * 32];
    __shared__ __attribute__((aligned(16))) float  sattn[16 * 16];
    __shared__ __attribute__((aligned(16))) __bf16 sco[16 * FF];
    __shared__ float sv[2 * FF];       // v1 | v2 (f32)
    __shared__ float sred[256];
    int t = threadIdx.x, w = t >> 5, lane = t & 31, lr = lane & 15, half = lane >> 4;
    int b = blockIdx.x;

    ((uint4*)sx)[t] = ((const uint4*)(meta_bf + (size_t)b * 16 * FF))[t];
    for (int i = t; i < FF * 16; i += 256)
        vT[(i >> 4) * 32 + 16 + (i & 15)] = bf_zero();
    __syncthreads();

    // 24 qkv tiles (mpa) + 8 value-path tiles (qia wv)
    for (int tt = w; tt < 32; tt += 8) {
        const __bf16* ar = sx + lr * FF;
        v8f acc = {};
        if (tt < 24) {
            int ct = tt;
            const __bf16* br = m_in_w_bf + (ct * 16 + lr) * FF;
#pragma unroll
            for (int kc = 0; kc < 4; ++kc)
                acc = wmma_bf16(load_row(ar, kc * 32), load_row(br, kc * 32), acc);
            int col16 = (ct & 7) * 16 + lr;
            float bias = m_in_b[ct * 16 + lr];
            if (ct < 8) {
#pragma unroll
                for (int j = 0; j < 8; ++j) qb[(j + 8 * half) * FF + col16] = f2bf(acc[j] + bias);
            } else if (ct < 16) {
#pragma unroll
                for (int j = 0; j < 8; ++j) kb[(j + 8 * half) * FF + col16] = f2bf(acc[j] + bias);
            } else {
#pragma unroll
                for (int j = 0; j < 8; ++j) vT[col16 * 32 + (j + 8 * half)] = f2bf(acc[j] + bias);
            }
        } else {
            int ct = tt - 24;
            const __bf16* br = q_in_w_bf + (2 * FF + ct * 16 + lr) * FF; // wv rows
#pragma unroll
            for (int kc = 0; kc < 4; ++kc)
                acc = wmma_bf16(load_row(ar, kc * 32), load_row(br, kc * 32), acc);
            int col = ct * 16 + lr;
            float bias = q_in_b[2 * FF + col];
#pragma unroll
            for (int j = 0; j < 8; ++j) st[(j + 8 * half) * FF + col] = f2bf(acc[j] + bias);
        }
    }
    __syncthreads();

    if (w == 0) {
        v8f acc = {};
#pragma unroll
        for (int kc = 0; kc < 4; ++kc)
            acc = wmma_bf16(load_row(qb + lr * FF, kc * 32),
                            load_row(kb + lr * FF, kc * 32), acc);
        const float scale = 0.08838834764831845f;
#pragma unroll
        for (int j = 0; j < 8; ++j) sattn[(j + 8 * half) * 16 + lr] = acc[j] * scale;
    }
    __syncthreads();

    if (t < 16) {
        float mx = -1e30f;
        for (int j = 0; j < 16; ++j) mx = fmaxf(mx, sattn[t * 16 + j]);
        float e[16], s = 0.0f;
        for (int j = 0; j < 16; ++j) { e[j] = __expf(sattn[t * 16 + j] - mx); s += e[j]; }
        float inv = 1.0f / s;
        for (int j = 0; j < 16; ++j) {
            attn_bf[t * 32 + j] = f2bf(e[j] * inv);
            attn_bf[t * 32 + 16 + j] = bf_zero();
        }
    }
    __syncthreads();

    {
        int ct = w;
        v8f acc = wmma_bf16(load_row(attn_bf + lr * 32, 0),
                            load_row(vT + (ct * 16 + lr) * 32, 0), (v8f){});
#pragma unroll
        for (int j = 0; j < 8; ++j)
            sco[(j + 8 * half) * FF + ct * 16 + lr] = f2bf(acc[j]);
    }
    __syncthreads();

    // v1 = mean_P(core @ m_out_w^T + b);  v2 = mean_P(st @ q_out_w^T + b)
    for (int tt = w; tt < 16; tt += 8) {
        int ct = tt & 7;
        bool isV2 = tt >= 8;
        const __bf16* ar = (isV2 ? st : sco) + lr * FF;
        const __bf16* ww = (isV2 ? q_out_w_bf : m_out_w_bf) + (ct * 16 + lr) * FF;
        const float* bb = isV2 ? q_out_b : m_out_b;
        v8f acc = {};
#pragma unroll
        for (int kc = 0; kc < 4; ++kc)
            acc = wmma_bf16(load_row(ar, kc * 32), load_row(ww, kc * 32), acc);
        float ps = 0.0f;
#pragma unroll
        for (int j = 0; j < 8; ++j) ps += acc[j];
        ps += __shfl_xor(ps, 16, 32);
        if (half == 0) {
            int col = ct * 16 + lr;
            sv[(isV2 ? FF : 0) + col] = ps * (1.0f / 16.0f) + bb[col];
        }
    }
    __syncthreads();

    // base[b] = query.w[0:128] + v1.w[128:256] + v2.w[256:384] + score_b
    float partial = 0.0f;
    if (t < FF) {
        partial = query[(size_t)b * DD + t] * score_w[t]
                + sv[t] * score_w[FF + t]
                + sv[FF + t] * score_w[2 * FF + t];
    }
    sred[t] = partial;
    __syncthreads();
    for (int s = 128; s > 0; s >>= 1) {
        if (t < s) sred[t] += sred[t + s];
        __syncthreads();
    }
    if (t == 0) base_out[b] = sred[0] + score_b[0];
}

// ---------------------------------------------------------------------------
// K5: logits[b, p*C + c] = mask ? (base[b] + cand_emb[b,p,c,:].w_c) : NEG
// One candidate per wave; float4 coalesced loads (512B per wave).
// ---------------------------------------------------------------------------
__global__ __launch_bounds__(256) void k_score(const float* __restrict__ cand,
                                               const int* __restrict__ clen,
                                               const float* __restrict__ score_w,
                                               const float* __restrict__ base_in,
                                               float* __restrict__ logits) {
    int t = threadIdx.x, w = t >> 5, lane = t & 31;
    long ci = (long)blockIdx.x * 8 + w;      // 0 .. B*P*C-1
    int b = (int)(ci >> 14);                 // / (P*C = 16384)
    int rem = (int)(ci & 16383);
    int p = rem >> 10;
    int c = rem & 1023;
    const float4* src = (const float4*)cand + ci * 32 + lane;
    const float4* wv  = (const float4*)(score_w + 384) + lane;
    float4 x = *src, y = *wv;
    float ps = x.x * y.x + x.y * y.y + x.z * y.z + x.w * y.w;
#pragma unroll
    for (int off = 16; off >= 1; off >>= 1) ps += __shfl_xor(ps, off, 32);
    if (lane == 0) {
        float s = ps + base_in[b];
        logits[ci] = (c < clen[b * PP + p]) ? s : NEGV;
    }
}

// ---------------------------------------------------------------------------
// K6: softmax over P*C = 16384 per b (logits are L2-resident, two passes)
// ---------------------------------------------------------------------------
__global__ __launch_bounds__(256) void k_softmax(const float* __restrict__ logits,
                                                 float* __restrict__ out) {
    __shared__ float sred[256];
    int b = blockIdx.x, t = threadIdx.x;
    const float* x = logits + (size_t)b * PP * CC;
    float mx = -1e30f;
    for (int i = t; i < PP * CC; i += 256) mx = fmaxf(mx, x[i]);
    sred[t] = mx;
    __syncthreads();
    for (int s = 128; s > 0; s >>= 1) {
        if (t < s) sred[t] = fmaxf(sred[t], sred[t + s]);
        __syncthreads();
    }
    float M = sred[0];
    __syncthreads();
    float sm = 0.0f;
    for (int i = t; i < PP * CC; i += 256) sm += __expf(x[i] - M);
    sred[t] = sm;
    __syncthreads();
    for (int s = 128; s > 0; s >>= 1) {
        if (t < s) sred[t] += sred[t + s];
        __syncthreads();
    }
    float inv = 1.0f / sred[0];
    for (int i = t; i < PP * CC; i += 256)
        out[(size_t)b * PP * CC + i] = __expf(x[i] - M) * inv;
}

// ---------------------------------------------------------------------------
extern "C" void kernel_launch(void* const* d_in, const int* in_sizes, int n_in,
                              void* d_out, int out_size, void* d_ws, size_t ws_size,
                              hipStream_t stream) {
    const float* query    = (const float*)d_in[0];
    const float* path_emb = (const float*)d_in[1];
    const int*   path_len = (const int*)  d_in[2];
    const float* cand_emb = (const float*)d_in[3];
    const int*   cand_len = (const int*)  d_in[4];
    const float* conv_w   = (const float*)d_in[5];
    const float* conv_b   = (const float*)d_in[6];
    const float* pia_in_w  = (const float*)d_in[7];
    const float* pia_in_b  = (const float*)d_in[8];
    const float* pia_out_w = (const float*)d_in[9];
    const float* pia_out_b = (const float*)d_in[10];
    const float* mpa_in_w  = (const float*)d_in[11];
    const float* mpa_in_b  = (const float*)d_in[12];
    const float* mpa_out_w = (const float*)d_in[13];
    const float* mpa_out_b = (const float*)d_in[14];
    const float* qia_in_w  = (const float*)d_in[15];
    const float* qia_in_b  = (const float*)d_in[16];
    const float* qia_out_w = (const float*)d_in[17];
    const float* qia_out_b = (const float*)d_in[18];
    const float* score_w   = (const float*)d_in[19];
    const float* score_b   = (const float*)d_in[20];

    const size_t NROWS = (size_t)BB * PP * II;          // 4096
    char* ws = (char*)d_ws;
    float*  mean_x  = (float*)ws;                                   // 2 MB f32
    __bf16* inst_bf = (__bf16*)(ws + NROWS * DD * 4);               // 1 MB bf16
    __bf16* meta_bf = (__bf16*)(ws + NROWS * DD * 4 + NROWS * DD * 2);      // 64 KB
    float*  base    = (float*)(ws + NROWS * DD * 4 + NROWS * DD * 2
                                  + (size_t)BB * PP * FF * 2);              // 64 B
    __bf16* wbf     = (__bf16*)(ws + NROWS * DD * 4 + NROWS * DD * 2
                                   + (size_t)BB * PP * FF * 2 + 256);
    float*  logits  = mean_x;   // reuse: mean_x dead after k_inst (1 MB)

    // bf16 weight regions (element offsets)
    const int SZ_SQ = FF * DD;       // 16384
    const int SZ_IN = 3 * FF * FF;   // 49152
    __bf16* conv_w_bf  = wbf;
    __bf16* pia_in_bf  = wbf + SZ_SQ;
    __bf16* pia_out_bf = pia_in_bf + SZ_IN;
    __bf16* mpa_in_bf  = pia_out_bf + SZ_SQ;
    __bf16* mpa_out_bf = mpa_in_bf + SZ_IN;
    __bf16* qia_in_bf  = mpa_out_bf + SZ_SQ;
    __bf16* qia_out_bf = qia_in_bf + SZ_IN;

    k_cvt<<<(SZ_SQ + 255) / 256, 256, 0, stream>>>(conv_w,   conv_w_bf,  SZ_SQ);
    k_cvt<<<(SZ_IN + 255) / 256, 256, 0, stream>>>(pia_in_w,  pia_in_bf,  SZ_IN);
    k_cvt<<<(SZ_SQ + 255) / 256, 256, 0, stream>>>(pia_out_w, pia_out_bf, SZ_SQ);
    k_cvt<<<(SZ_IN + 255) / 256, 256, 0, stream>>>(mpa_in_w,  mpa_in_bf,  SZ_IN);
    k_cvt<<<(SZ_SQ + 255) / 256, 256, 0, stream>>>(mpa_out_w, mpa_out_bf, SZ_SQ);
    k_cvt<<<(SZ_IN + 255) / 256, 256, 0, stream>>>(qia_in_w,  qia_in_bf,  SZ_IN);
    k_cvt<<<(SZ_SQ + 255) / 256, 256, 0, stream>>>(qia_out_w, qia_out_bf, SZ_SQ);

    k_mean<<<BB * PP * II, 256, 0, stream>>>(path_emb, path_len, mean_x);
    k_inst<<<(BB * PP * II) / 16, 256, 0, stream>>>(mean_x, conv_w_bf, conv_b, inst_bf);
    k_pia<<<BB * PP, 256, 0, stream>>>(inst_bf, pia_in_bf, pia_in_b,
                                       pia_out_bf, pia_out_b, meta_bf);
    k_mpa<<<BB, 256, 0, stream>>>(meta_bf, query,
                                  mpa_in_bf, mpa_in_b, mpa_out_bf, mpa_out_b,
                                  qia_in_bf, qia_in_b, qia_out_bf, qia_out_b,
                                  score_w, score_b, base);
    k_score<<<(BB * PP * CC) / 8, 256, 0, stream>>>(cand_emb, cand_len, score_w, base, logits);
    k_softmax<<<BB, 256, 0, stream>>>(logits, (float*)d_out);
}
